// PatchNet_ms_conv_66855460929919
// MI455X (gfx1250) — matrix operand order repaired and checked by reading, not compiled, not run on hardware
//
#include <hip/hip_runtime.h>
#include <hip/hip_bf16.h>
#include <math.h>

typedef _Float16 v8h  __attribute__((ext_vector_type(8)));
typedef _Float16 v16h __attribute__((ext_vector_type(16)));
typedef float    v8f  __attribute__((ext_vector_type(8)));

union AB16 { v16h v; v8h h[2]; };

__device__ __forceinline__ float gelu_exact(float x) {
    return 0.5f * x * (1.0f + erff(x * 0.70710678118654752f));
}

// ---------------------------------------------------------------------------
// Kernel 1: predictor.  grid = (49 tiles, 64 images), block = 256 (8 waves).
// Computes s0 = softmax(gelu(conv2(gelu(conv1(x)))))[ch0] for a 16x16 tile.
// conv1 (128->2, 3x3) done via v_wmma_f32_16x16x32_f16:
//   M = 16 pixels of the 18x18 (halo-1) conv1-output region
//   N = 16 (out-channels 0,1 real; 2..15 ignored)
//   K = 1152 = 4 channel-groups * 9 taps * 32 channels, chunked by 32
// ---------------------------------------------------------------------------
__global__ void __launch_bounds__(256)
predictor_kernel(const float* __restrict__ x,
                 const float* __restrict__ w1, const float* __restrict__ b1,
                 const float* __restrict__ w2, const float* __restrict__ b2,
                 float* __restrict__ s0out)
{
    const int tid  = threadIdx.x;
    const int lane = tid & 31;
    const int wave = tid >> 5;

    const int n  = blockIdx.y;            // image 0..63
    const int th = blockIdx.x / 7;
    const int tw = blockIdx.x % 7;
    const int h0 = th * 16;
    const int w0 = tw * 16;

    // xs[r][c][cc] : 20x20 halo tile of one 32-channel group, f16, channel-minor
    __shared__ __align__(16) _Float16 xs[20 * 20 * 32];      // 25.6 KB
    // bpack[(g*9+j)*4 + half*2 + n][16] : prepacked B fragments
    __shared__ __align__(16) _Float16 bpack[4 * 9 * 4 * 16]; //  4.6 KB
    // gelu(conv1) on the 18x18 region, 2 channels
    __shared__ float y1s[2 * 324];                           //  2.6 KB

    // ---- prepack B fragments (K order: tap-major, channel-minor) ----------
    for (int idx = tid; idx < 4 * 9 * 4 * 16; idx += 256) {
        int i  = idx & 15;        // half index within fragment -> channel lsb
        int e  = idx >> 4;
        int nn = e & 1;           // out channel
        int hh = (e >> 1) & 1;    // vector half b.h[hh] (cc 0..15 vs 16..31)
        int jg = e >> 2;          // 0..35
        int j  = jg % 9;          // tap
        int g  = jg / 9;          // channel group
        int ci = g * 32 + hh * 16 + i;
        bpack[idx] = (_Float16)w1[(size_t)(nn * 128 + ci) * 9 + j];
    }

    v8f acc[3] = {};

    // per-lane A-fragment constants
    const int m   = lane & 15;             // pixel-in-tile (M)
    const int ccA = (lane < 16) ? 0 : 8;   // first channel run for this lane

    for (int g = 0; g < 4; ++g) {
        __syncthreads();   // previous group fully consumed (also covers bpack on g==0)
        // ---- stage x[g*32 .. g*32+31] 20x20 halo tile into LDS as f16 -----
        for (int i = tid; i < 12800; i += 256) {
            int cc  = i / 400;
            int rem = i - cc * 400;
            int r   = rem / 20;
            int c   = rem - r * 20;
            int gh  = h0 - 2 + r;
            int gw  = w0 - 2 + c;
            float v = 0.0f;
            if ((unsigned)gh < 112u && (unsigned)gw < 112u)
                v = x[(size_t)(n * 128 + g * 32 + cc) * 12544 + gh * 112 + gw];
            xs[(r * 20 + c) * 32 + cc] = (_Float16)v;
        }
        __syncthreads();

        // ---- WMMA accumulate: 21 M-tiles over 8 waves, 9 taps each --------
        #pragma unroll
        for (int tt = 0; tt < 3; ++tt) {
            const int tile = wave + tt * 8;
            if (tile < 21) {                       // wave-uniform branch
                int p  = tile * 16 + m;
                int pp = (p < 324) ? p : 0;        // pad lanes read pixel 0 (finite garbage)
                int pr = pp / 18;
                int pc = pp - pr * 18;
                #pragma unroll
                for (int j = 0; j < 9; ++j) {
                    const int kh = j / 3, kw = j - kh * 3;
                    const _Float16* ap =
                        &xs[((pr + kh) * 20 + (pc + kw)) * 32 + ccA];
                    AB16 a, b;
                    a.h[0] = *(const v8h*)ap;          // cc {0-7 | 8-15}
                    a.h[1] = *(const v8h*)(ap + 16);   // cc {16-23 | 24-31}
                    const _Float16* bp =
                        &bpack[(((g * 9 + j) * 4) + (m & 1)) * 16 + ((lane >> 4) << 3)];
                    b.h[0] = *(const v8h*)bp;
                    b.h[1] = *(const v8h*)(bp + 32);
                    acc[tt] = __builtin_amdgcn_wmma_f32_16x16x32_f16(
                        false, a.v, false, b.v, (short)0, acc[tt], false, false);
                }
            }
        }
    }

    // ---- bias + GELU, scatter D (M=pixel, N=channel) into y1s -------------
    {
        const int nsel = lane & 15;
        if (nsel < 2) {
            const float bias = b1[nsel];
            #pragma unroll
            for (int tt = 0; tt < 3; ++tt) {
                const int tile = wave + tt * 8;
                if (tile < 21) {
                    #pragma unroll
                    for (int v = 0; v < 8; ++v) {
                        int mm = v + ((lane >> 4) << 3);   // M = v (+8 for hi lanes)
                        int p  = tile * 16 + mm;
                        if (p < 324)
                            y1s[nsel * 324 + p] = gelu_exact(acc[tt][v] + bias);
                    }
                }
            }
        }
    }
    __syncthreads();

    // ---- conv2 (2->2, 3x3) + GELU + softmax(dim=channel) ------------------
    {
        const int i  = tid >> 4;     // 0..15 row
        const int jx = tid & 15;     // 0..15 col
        const int r  = i + 1;        // coords in 18x18 region
        const int c  = jx + 1;
        float z0 = b2[0], z1 = b2[1];
        #pragma unroll
        for (int ci2 = 0; ci2 < 2; ++ci2) {
            #pragma unroll
            for (int kh = 0; kh < 3; ++kh) {
                #pragma unroll
                for (int kw = 0; kw < 3; ++kw) {
                    float v = y1s[ci2 * 324 + (r - 1 + kh) * 18 + (c - 1 + kw)];
                    z0 += v * w2[(0 * 2 + ci2) * 9 + kh * 3 + kw];
                    z1 += v * w2[(1 * 2 + ci2) * 9 + kh * 3 + kw];
                }
            }
        }
        float g0 = gelu_exact(z0);
        float g1 = gelu_exact(z1);
        float s0 = 1.0f / (1.0f + __expf(g1 - g0));   // softmax over 2 channels
        s0out[(size_t)n * 12544 + (h0 + i) * 112 + (w0 + jx)] = s0;
    }
}

// ---------------------------------------------------------------------------
// Kernel 2: xa = s0*x1 + (1-s0)*x2 with '(b t) c h w -> b c t h w' fold.
// Pure streaming, float4 vectorized. 12,845,056 float4s total.
// ---------------------------------------------------------------------------
__global__ void __launch_bounds__(256)
combine_kernel(const float* __restrict__ x, const float* __restrict__ s0,
               float* __restrict__ out)
{
    const int idx = blockIdx.x * 256 + threadIdx.x;   // float4 index
    if (idx >= 12845056) return;
    const int hw4   = idx % 3136;          // 12544/4
    const int plane = idx / 3136;          // ((b*64 + c)*16 + t)
    const int t = plane & 15;
    const int c = (plane >> 4) & 63;
    const int b = plane >> 10;
    const int img = b * 16 + t;

    const float4* x1p = (const float4*)x + (size_t)(img * 128 + c) * 3136 + hw4;
    const float4* x2p = x1p + (size_t)64 * 3136;
    const float4  s   = ((const float4*)s0)[(size_t)img * 3136 + hw4];
    const float4  a   = *x1p;
    const float4  d   = *x2p;

    float4 r;
    r.x = s.x * a.x + (1.0f - s.x) * d.x;
    r.y = s.y * a.y + (1.0f - s.y) * d.y;
    r.z = s.z * a.z + (1.0f - s.z) * d.z;
    r.w = s.w * a.w + (1.0f - s.w) * d.w;
    ((float4*)out)[idx] = r;
}

extern "C" void kernel_launch(void* const* d_in, const int* in_sizes, int n_in,
                              void* d_out, int out_size, void* d_ws, size_t ws_size,
                              hipStream_t stream)
{
    const float* x  = (const float*)d_in[0];
    const float* w1 = (const float*)d_in[1];
    const float* b1 = (const float*)d_in[2];
    const float* w2 = (const float*)d_in[3];
    const float* b2 = (const float*)d_in[4];
    float* out = (float*)d_out;
    float* s0  = (float*)d_ws;             // 64*12544 floats = 3.2 MB scratch

    dim3 grid1(49, 64);
    predictor_kernel<<<grid1, 256, 0, stream>>>(x, w1, b1, w2, b2, s0);

    const int n4 = 12845056;               // 51,380,224 / 4
    combine_kernel<<<n4 / 256, 256, 0, stream>>>(x, s0, out);
}
